// Sentence_Function_Interaction_34918084116967
// MI455X (gfx1250) — matrix operand order repaired and checked by reading, not compiled, not run on hardware
//
#include <hip/hip_runtime.h>
#include <math.h>

#define NS 16384
#define NF 4096
#define DIM 768
#define CHUNK 1024
#define LOG2E 1.4426950408889634f

typedef __attribute__((ext_vector_type(16))) __bf16 bf16x16;
typedef __attribute__((ext_vector_type(8)))  __bf16 bf16x8;
typedef __attribute__((ext_vector_type(4)))  __bf16 bf16x4;
typedef __attribute__((ext_vector_type(8)))  float   v8f;

union bfrag { bf16x16 v; bf16x8 h[2]; };

// A-matrix 16x32 bf16 fragment: lane (l&15)=row M; lanes>=16 -> K base +16.
// p must already point at: base + (lane&15)*ld + ((lane>>4)*16)
// elements [0..15] = K offset +0..15 (contiguous)
__device__ __forceinline__ bf16x16 frag_ld_a(const __bf16* p) {
  bfrag u;
  u.h[0] = *(const bf16x8*)(p);
  u.h[1] = *(const bf16x8*)(p + 8);
  return u.v;
}

// B-matrix 32x16 bf16 fragment: lane (l&15)=col N; lanes 0-15 hold K=0..15,
// lanes 16-31 hold K=16..31 (contiguous 16 elements per lane).
// p must already point at: base + (lane&15)*ld + ((lane>>4)*16)
__device__ __forceinline__ bf16x16 frag_ld_b(const __bf16* p) {
  bfrag u;
  u.h[0] = *(const bf16x8*)(p);
  u.h[1] = *(const bf16x8*)(p + 8);
  return u.v;
}

__device__ __forceinline__ v8f wmma_bf16(bf16x16 a, bf16x16 b, v8f c) {
  return __builtin_amdgcn_wmma_f32_16x16x32_bf16(false, a, false, b, (short)0, c,
                                                 false, false);
}

#define SCHED_FENCE() __builtin_amdgcn_sched_barrier(0)

// ---------------------------------------------------------------- init / cvt
__global__ void cvt_kernel(const float* __restrict__ src, float* __restrict__ dst32,
                           __bf16* __restrict__ dstbf, int n) {
  for (int i = blockIdx.x * blockDim.x + threadIdx.x; i < n;
       i += gridDim.x * blockDim.x) {
    float v = src[i];
    if (dst32) dst32[i] = v;
    dstbf[i] = (__bf16)v;
  }
}

// ------------------------------------------------ value projection (transposed)
// valt[d][kv] = sum_k Wv[d][k] * hid[kv][k] + bv[d]      (valt is [DIM x NKV])
// Each wave: 16 d-rows x 64 kv-cols (4 WMMA n-tiles); ping-pong K pipeline with
// sched fences so loads stay one phase ahead of the WMMA groups.
__global__ void value_proj_kernel(const __bf16* __restrict__ wbf,   // [DIM x DIM]
                                  const float*  __restrict__ bias,  // [DIM]
                                  const __bf16* __restrict__ hidbf, // [NKV x DIM]
                                  __bf16* __restrict__ valt,        // [DIM x NKV]
                                  int NKV) {
  int wave = threadIdx.x >> 5, lane = threadIdx.x & 31;
  int nkvb = NKV >> 6;
  int id   = blockIdx.x * 8 + wave;
  int d0   = (id / nkvb) * 16;
  int kv0  = (id % nkvb) * 64;
  int r    = lane & 15;
  int h    = lane >> 4;
  int koffA = h * 16, koffB = h * 16;

  v8f acc[4] = {};
  const __bf16* arow = wbf + (size_t)(d0 + r) * DIM + koffA;
  const __bf16* brow = hidbf + (size_t)(kv0 + r) * DIM + koffB;

  bf16x16 aA = frag_ld_a(arow);
  bf16x16 aB, bA[4], bB[4];
#pragma unroll
  for (int t = 0; t < 4; ++t) bA[t] = frag_ld_b(brow + (size_t)(t * 16) * DIM);

#pragma unroll
  for (int k0 = 0; k0 < DIM; k0 += 64) {
    aB = frag_ld_a(arow + k0 + 32);
#pragma unroll
    for (int t = 0; t < 4; ++t)
      bB[t] = frag_ld_b(brow + (size_t)(t * 16) * DIM + k0 + 32);
    SCHED_FENCE();
#pragma unroll
    for (int t = 0; t < 4; ++t) acc[t] = wmma_bf16(aA, bA[t], acc[t]);
    SCHED_FENCE();
    if (k0 + 64 < DIM) {
      aA = frag_ld_a(arow + k0 + 64);
#pragma unroll
      for (int t = 0; t < 4; ++t)
        bA[t] = frag_ld_b(brow + (size_t)(t * 16) * DIM + k0 + 64);
    }
    SCHED_FENCE();
#pragma unroll
    for (int t = 0; t < 4; ++t) acc[t] = wmma_bf16(aB, bB[t], acc[t]);
    SCHED_FENCE();
  }
#pragma unroll
  for (int v = 0; v < 8; ++v) {
    int dd = d0 + v + 8 * h;
    float bb = bias[dd];
#pragma unroll
    for (int t = 0; t < 4; ++t)
      valt[(size_t)dd * NKV + kv0 + t * 16 + r] = (__bf16)(acc[t][v] + bb);
  }
}

// ---------------------------------------------- pass 1: logits z + softmax stats
// One workgroup = 16 Q rows; wave w owns 64 KV cols (4 tiles) per step of 512.
__global__ void attn_scores_kernel(const __bf16* __restrict__ Qbf,  // [NQ x DIM]
                                   const __bf16* __restrict__ Kbf,  // [NK x DIM]
                                   float* __restrict__ z,           // [NQ x NK]
                                   float* __restrict__ smax,        // [NQ]
                                   float* __restrict__ sinv,        // [NQ] = 1/L
                                   int NK, float scale) {
  __shared__ __bf16 qs[16 * DIM];       // 24 KB Q tile
  __shared__ float red[8][16][2];
  int tid = threadIdx.x;
  int row0 = blockIdx.x * 16;
  {
    const bf16x8* s = (const bf16x8*)(Qbf + (size_t)row0 * DIM);
    bf16x8* d = (bf16x8*)qs;
    for (int i = tid; i < 16 * DIM / 8; i += 256) d[i] = s[i];
  }
  __syncthreads();

  int wave = tid >> 5, lane = tid & 31;
  int r = lane & 15, h = lane >> 4;
  int koffA = h * 16, koffB = h * 16;
  float rmax[8], rsum[8];
#pragma unroll
  for (int v = 0; v < 8; ++v) { rmax[v] = -__builtin_inff(); rsum[v] = 0.0f; }

  const __bf16* qrow = qs + r * DIM + koffA;

  for (int n0 = wave * 64; n0 < NK; n0 += 512) {
    v8f acc[4] = {};
    const __bf16* kb0 = Kbf + (size_t)(n0 + r) * DIM + koffB;

    bf16x16 aA = frag_ld_a(qrow);
    bf16x16 aB, bA[4], bB[4];
#pragma unroll
    for (int t = 0; t < 4; ++t) bA[t] = frag_ld_b(kb0 + (size_t)(t * 16) * DIM);

#pragma unroll
    for (int k0 = 0; k0 < DIM; k0 += 64) {
      aB = frag_ld_a(qrow + k0 + 32);
#pragma unroll
      for (int t = 0; t < 4; ++t)
        bB[t] = frag_ld_b(kb0 + (size_t)(t * 16) * DIM + k0 + 32);
      SCHED_FENCE();
#pragma unroll
      for (int t = 0; t < 4; ++t) acc[t] = wmma_bf16(aA, bA[t], acc[t]);
      SCHED_FENCE();
      if (k0 + 64 < DIM) {
        aA = frag_ld_a(qrow + k0 + 64);
#pragma unroll
        for (int t = 0; t < 4; ++t)
          bA[t] = frag_ld_b(kb0 + (size_t)(t * 16) * DIM + k0 + 64);
      }
      SCHED_FENCE();
#pragma unroll
      for (int t = 0; t < 4; ++t) acc[t] = wmma_bf16(aB, bB[t], acc[t]);
      SCHED_FENCE();
    }

    // scale + store logits (C/D layout: element v -> row v+8h, col lane&15)
#pragma unroll
    for (int t = 0; t < 4; ++t) {
      float* zp = z + (size_t)row0 * NK + n0 + t * 16;
#pragma unroll
      for (int v = 0; v < 8; ++v) {
        float s = acc[t][v] * scale;
        acc[t][v] = s;
        zp[(size_t)(v + 8 * h) * NK + r] = s;
      }
    }
    // online softmax over the 64 cols this wave produced
#pragma unroll
    for (int v = 0; v < 8; ++v) {
      float mx = fmaxf(fmaxf(acc[0][v], acc[1][v]), fmaxf(acc[2][v], acc[3][v]));
      mx = fmaxf(mx, __shfl_xor(mx, 1, 32));
      mx = fmaxf(mx, __shfl_xor(mx, 2, 32));
      mx = fmaxf(mx, __shfl_xor(mx, 4, 32));
      mx = fmaxf(mx, __shfl_xor(mx, 8, 32));
      float nm = fmaxf(rmax[v], mx);
      float e = exp2f((acc[0][v] - nm) * LOG2E) + exp2f((acc[1][v] - nm) * LOG2E) +
                exp2f((acc[2][v] - nm) * LOG2E) + exp2f((acc[3][v] - nm) * LOG2E);
      e += __shfl_xor(e, 1, 32);
      e += __shfl_xor(e, 2, 32);
      e += __shfl_xor(e, 4, 32);
      e += __shfl_xor(e, 8, 32);
      rsum[v] = rsum[v] * exp2f((rmax[v] - nm) * LOG2E) + e;
      rmax[v] = nm;
    }
  }
  if ((lane & 15) == 0) {
#pragma unroll
    for (int v = 0; v < 8; ++v) {
      red[wave][8 * h + v][0] = rmax[v];
      red[wave][8 * h + v][1] = rsum[v];
    }
  }
  __syncthreads();
  if (tid < 16) {
    float M = -__builtin_inff();
    for (int w = 0; w < 8; ++w) M = fmaxf(M, red[w][tid][0]);
    float L = 0.0f;
    for (int w = 0; w < 8; ++w)
      L += red[w][tid][1] * exp2f((red[w][tid][0] - M) * LOG2E);
    smax[row0 + tid] = M;
    sinv[row0 + tid] = 1.0f / L;
  }
}

// ----------------------------- pass 2: O = P @ V, residual add, bf16 shadow out
// One workgroup = 16 Q rows x full DIM; wave w owns cols [96w, 96w+96).
// P chunk (16 x CHUNK, bf16) built once in LDS, shared by all 8 waves.
__global__ void attn_pv_kernel(const float* __restrict__ z,     // [NQ x NK]
                               const float* __restrict__ smax,
                               const float* __restrict__ sinv,
                               const __bf16* __restrict__ valt, // [DIM x NK]
                               float* __restrict__ hid,         // [NQ x DIM] in/out
                               __bf16* __restrict__ hidbf,      // [NQ x DIM] out
                               int NK) {
  __shared__ __bf16 pbuf[16 * CHUNK];   // 32 KB
  int tid = threadIdx.x;
  int row0 = blockIdx.x * 16;
  int wave = tid >> 5, lane = tid & 31;
  int r = lane & 15, h = lane >> 4;
  int koffA = h * 16, koffB = h * 16;
  int col0 = wave * 96;

  v8f acc[6] = {};
  for (int kb = 0; kb < NK; kb += CHUNK) {
    __syncthreads();
#pragma unroll
    for (int j = 0; j < 16; ++j) {   // stage P = exp(z-M)*invL as bf16
      float M = smax[row0 + j], iL = sinv[row0 + j];
      float4 zv = *(const float4*)(z + (size_t)(row0 + j) * NK + kb + tid * 4);
      bf16x4 pv;
      pv[0] = (__bf16)(exp2f((zv.x - M) * LOG2E) * iL);
      pv[1] = (__bf16)(exp2f((zv.y - M) * LOG2E) * iL);
      pv[2] = (__bf16)(exp2f((zv.z - M) * LOG2E) * iL);
      pv[3] = (__bf16)(exp2f((zv.w - M) * LOG2E) * iL);
      *(bf16x4*)(pbuf + j * CHUNK + tid * 4) = pv;
    }
    __syncthreads();

    // warm L2 with next z chunk while this chunk's WMMAs run (HBM stream)
    if (kb + CHUNK < NK) {
#pragma unroll
      for (int j = 0; j < 16; ++j)
        __builtin_prefetch(z + (size_t)(row0 + j) * NK + kb + CHUNK + tid * 4, 0, 0);
    }

    const __bf16* prow = pbuf + r * CHUNK + koffA;
    const __bf16* vb = valt + (size_t)(col0 + r) * NK + kb + koffB;

    bf16x16 aA = frag_ld_a(prow);
    bf16x16 aB, bA[6], bB[6];
#pragma unroll
    for (int t = 0; t < 6; ++t) bA[t] = frag_ld_b(vb + (size_t)(t * 16) * NK);

#pragma unroll 1
    for (int kk = 0; kk < CHUNK; kk += 64) {
      aB = frag_ld_a(prow + kk + 32);
#pragma unroll
      for (int t = 0; t < 6; ++t)
        bB[t] = frag_ld_b(vb + (size_t)(t * 16) * NK + kk + 32);
      SCHED_FENCE();
#pragma unroll
      for (int t = 0; t < 6; ++t) acc[t] = wmma_bf16(aA, bA[t], acc[t]);
      SCHED_FENCE();
      if (kk + 64 < CHUNK) {
        aA = frag_ld_a(prow + kk + 64);
#pragma unroll
        for (int t = 0; t < 6; ++t)
          bA[t] = frag_ld_b(vb + (size_t)(t * 16) * NK + kk + 64);
      }
      SCHED_FENCE();
#pragma unroll
      for (int t = 0; t < 6; ++t) acc[t] = wmma_bf16(aB, bB[t], acc[t]);
      SCHED_FENCE();
    }
  }
#pragma unroll
  for (int t = 0; t < 6; ++t) {
#pragma unroll
    for (int v = 0; v < 8; ++v) {
      size_t idx = (size_t)(row0 + v + 8 * h) * DIM + col0 + t * 16 + r;
      float o = hid[idx] + acc[t][v];
      hid[idx] = o;
      hidbf[idx] = (__bf16)o;
    }
  }
}

// -------------------------------------------------------------------- host
extern "C" void kernel_launch(void* const* d_in, const int* in_sizes, int n_in,
                              void* d_out, int out_size, void* d_ws, size_t ws_size,
                              hipStream_t stream) {
  const float* sents_emb = (const float*)d_in[0];
  const float* func_emb  = (const float*)d_in[1];
  const float* Wv_f2s    = (const float*)d_in[2];
  const float* bv_f2s    = (const float*)d_in[3];
  const float* Wv_s2f    = (const float*)d_in[4];
  const float* bv_s2f    = (const float*)d_in[5];

  float* out = (float*)d_out;
  float* sents_hid = out;                                  // [NS x DIM]
  float* func_hid  = out + (size_t)NS * DIM;               // [NF x DIM]
  float* z_out     = func_hid + (size_t)NF * DIM;          // [NS x NF]

  char* p = (char*)d_ws;
  auto take = [&](size_t bytes) {
    char* q = p;
    p += (bytes + 255) & ~(size_t)255;
    return q;
  };
  __bf16* sents_bf = (__bf16*)take((size_t)NS * DIM * 2);  // 25.2 MB
  __bf16* func_bf  = (__bf16*)take((size_t)NF * DIM * 2);  //  6.3 MB
  __bf16* wvf_bf   = (__bf16*)take((size_t)DIM * DIM * 2);
  __bf16* wvs_bf   = (__bf16*)take((size_t)DIM * DIM * 2);
  __bf16* valt     = (__bf16*)take((size_t)DIM * NS * 2);  // 25.2 MB (reused)
  float*  z2       = (float*) take((size_t)NF * NS * 4);   // 268 MB
  float*  s_max    = (float*) take((size_t)NS * 4);
  float*  s_inv    = (float*) take((size_t)NS * 4);
  float*  f_max    = (float*) take((size_t)NF * 4);
  float*  f_inv    = (float*) take((size_t)NF * 4);

  const float scale = 1.0f / sqrtf((float)DIM);

  // init: fp32 masters into d_out, bf16 shadows + weight conversion into ws
  cvt_kernel<<<2048, 256, 0, stream>>>(sents_emb, sents_hid, sents_bf, NS * DIM);
  cvt_kernel<<<512,  256, 0, stream>>>(func_emb,  func_hid,  func_bf,  NF * DIM);
  cvt_kernel<<<256,  256, 0, stream>>>(Wv_f2s, nullptr, wvf_bf, DIM * DIM);
  cvt_kernel<<<256,  256, 0, stream>>>(Wv_s2f, nullptr, wvs_bf, DIM * DIM);

  for (int it = 0; it < 4; ++it) {
    // func2sent: V = Wv_f2s(func), z into d_out (final iteration's z is output)
    value_proj_kernel<<<(48 * (NF / 64)) / 8, 256, 0, stream>>>(
        wvf_bf, bv_f2s, func_bf, valt, NF);
    attn_scores_kernel<<<NS / 16, 256, 0, stream>>>(
        sents_bf, func_bf, z_out, s_max, s_inv, NF, scale);
    attn_pv_kernel<<<NS / 16, 256, 0, stream>>>(
        z_out, s_max, s_inv, valt, sents_hid, sents_bf, NF);
    // sent2func: V = Wv_s2f(updated sents)
    value_proj_kernel<<<(48 * (NS / 64)) / 8, 256, 0, stream>>>(
        wvs_bf, bv_s2f, sents_bf, valt, NS);
    attn_scores_kernel<<<NF / 16, 256, 0, stream>>>(
        func_bf, sents_bf, z2, f_max, f_inv, NS, scale);
    attn_pv_kernel<<<NF / 16, 256, 0, stream>>>(
        z2, f_max, f_inv, valt, func_hid, func_bf, NS);
  }
}
